// ECInteract_85753317032171
// MI455X (gfx1250) — matrix operand order repaired and checked by reading, not compiled, not run on hardware
//
#include <hip/hip_runtime.h>
#include <hip/hip_bf16.h>

typedef __attribute__((ext_vector_type(16))) _Float16 v16h;
typedef __attribute__((ext_vector_type(8)))  _Float16 v8h;
typedef __attribute__((ext_vector_type(8)))  float    v8f;

// Problem constants
#define BB 4
#define CC 64
#define HH 128
#define WW 256
#define FH 128   // cube face size h

// ---------------------------------------------------------------------------
// Kernel 0: swizzle w_fusion [64,64,3,3] f32 -> WMMA B-fragment order, f16.
// Layout: wfrag[kt][nt][lane][e], kt=0..17 (K tiles of 32), nt=0..3 (16 couts),
// lane 0..31, e 0..15.  n = nt*16 + (lane&15); K = kt*32 + (e>>3)*16 + (lane>>4)*8 + (e&7),
// with K decomposed as K = (3*dy+dx)*64 + c_in.
// ---------------------------------------------------------------------------
__global__ __launch_bounds__(256)
void prep_weights(const float* __restrict__ w, _Float16* __restrict__ wfrag) {
    int idx = blockIdx.x * 256 + threadIdx.x;
    if (idx >= 18 * 4 * 32 * 16) return;
    int e    =  idx        & 15;
    int lane = (idx >> 4)  & 31;
    int nt   = (idx >> 9)  & 3;
    int kt   =  idx >> 11;
    int n = nt * 16 + (lane & 15);
    int k = kt * 32 + (e >> 3) * 16 + (lane >> 4) * 8 + (e & 7);
    int dydx = k >> 6;          // 0..8
    int c    = k & 63;
    int dy = dydx / 3, dx = dydx - 3 * dy;
    wfrag[idx] = (_Float16)w[((n * CC + c) * 3 + dy) * 3 + dx];
}

// ---------------------------------------------------------------------------
// Kernel 1: cube -> equirect bilinear resample, writes aux as f16 [B,C,H,W].
// Geometry depends only on (y,x); each thread computes it once and loops over
// 16 of the 256 (b,c) planes (grid.y selects the group).
// ---------------------------------------------------------------------------
__global__ __launch_bounds__(256)
void c2e_kernel(const float* __restrict__ fb, const float* __restrict__ fd,
                const float* __restrict__ ff, const float* __restrict__ fl,
                const float* __restrict__ fr, const float* __restrict__ fu,
                _Float16* __restrict__ aux) {
    const int x   = threadIdx.x;      // 0..255
    const int y   = blockIdx.x;       // 0..127
    const int grp = blockIdx.y;       // 0..15 -> (b,c) planes grp*16..+16
    const float PI = 3.14159265358979f;
    float lat = (0.5f - (y + 0.5f) / (float)HH) * PI;
    float lon = (2.0f * (x + 0.5f) / (float)WW - 1.0f) * PI;
    float cl = cosf(lat);
    float vx = cl * sinf(lon);
    float vy = sinf(lat);
    float vz = cl * cosf(lon);
    float ax = fabsf(vx), ay = fabsf(vy), az = fabsf(vz);
    int face; float den, a, bc;
    if (az >= ax && az >= ay) {
        if (vz > 0.f) { face = 2; den =  vz; a =  vx / den; bc =  vy / den; }  // front
        else          { face = 0; den = -vz; a = -vx / den; bc =  vy / den; }  // back
    } else if (ax >= ay) {
        if (vx > 0.f) { face = 4; den =  vx; a = -vz / den; bc =  vy / den; }  // right
        else          { face = 3; den = -vx; a =  vz / den; bc =  vy / den; }  // left
    } else {
        if (vy > 0.f) { face = 5; den =  vy; a =  vx / den; bc =  vz / den; }  // up
        else          { face = 1; den = -vy; a =  vx / den; bc = -vz / den; }  // down
    }
    float uu = fminf(fmaxf((a + 1.0f) * 0.5f * (FH - 1), 0.f), (float)(FH - 1));
    float vv = fminf(fmaxf((1.0f - bc) * 0.5f * (FH - 1), 0.f), (float)(FH - 1));
    int x0 = (int)floorf(uu), y0 = (int)floorf(vv);
    int x1 = min(x0 + 1, FH - 1), y1 = min(y0 + 1, FH - 1);
    float fx = uu - (float)x0, fy = vv - (float)y0;
    float w00 = (1.f - fx) * (1.f - fy), w01 = fx * (1.f - fy);
    float w10 = (1.f - fx) * fy,         w11 = fx * fy;
    const float* fp;
    switch (face) {
        case 0: fp = fb; break; case 1: fp = fd; break; case 2: fp = ff; break;
        case 3: fp = fl; break; case 4: fp = fr; break; default: fp = fu; break;
    }
    #pragma unroll 4
    for (int i = 0; i < 16; ++i) {
        int bci = grp * 16 + i;                       // flat b*C + c
        const float* base = fp + (size_t)bci * FH * FH;
        float v = base[y0 * FH + x0] * w00 + base[y0 * FH + x1] * w01
                + base[y1 * FH + x0] * w10 + base[y1 * FH + x1] * w11;
        aux[((size_t)bci * HH + y) * WW + x] = (_Float16)v;
    }
}

// ---------------------------------------------------------------------------
// Kernel 2: fused 3x3 conv (implicit GEMM via v_wmma_f32_16x16x32_f16)
//           + bias + relu + 1x1-mask partial reduction.
// Block = 128 threads (4 wave32s). Block tile: 64 pixels (one row segment)
// x all 64 output channels. Each wave: 16 pixels x 4 WMMA N-tiles.
// GEMM: M=pixels, N=c_out, K = (3*dy+dx)*64 + c_in  (576 total, 18 tiles).
// If maux_out != null, stores relu result (f16); always stores the per-pixel
// partial  sum_c w_mask[wm_off + c] * relu(conv)[c]  (m_main never hits HBM).
// ---------------------------------------------------------------------------
template <typename T>
__global__ __launch_bounds__(128)
void conv_fusion_wmma(const T* __restrict__ in,            // [B,64,H,W]
                      const _Float16* __restrict__ wfrag,  // [18][4][32][16]
                      const float* __restrict__ bias,      // [64]
                      const float* __restrict__ wmask,     // [128]
                      int wm_off,
                      _Float16* __restrict__ maux_out,     // [B,64,H,W] or null
                      float* __restrict__ partial_out) {   // [B,H,W]
    __shared__ __align__(16) _Float16 s_in[3 * 66 * CC];   // [row][xx][c_in]
    const int xtile = blockIdx.x;        // 0..3
    const int y     = blockIdx.y;        // 0..127
    const int b     = blockIdx.z;        // 0..3
    const int xbase = xtile * 64;
    const int tid   = threadIdx.x;

    // Stage input rows y-1..y+1, x in [xbase-1, xbase+65), f32->f16, zero-pad.
    for (int e = tid; e < CC * 3 * 66; e += 128) {
        int xx  = e % 66;
        int row = (e / 66) % 3;
        int c   = e / (3 * 66);
        int gx = xbase - 1 + xx;
        int gy = y - 1 + row;
        float v = 0.0f;
        if (gx >= 0 && gx < WW && gy >= 0 && gy < HH)
            v = (float)in[(((size_t)b * CC + c) * HH + gy) * WW + gx];
        s_in[(row * 66 + xx) * CC + c] = (_Float16)v;
    }
    __syncthreads();

    const int wave = tid >> 5;
    const int lane = tid & 31;
    const int lo = lane & 15, hi = lane >> 4;
    v8f acc0 = {}, acc1 = {}, acc2 = {}, acc3 = {};

    for (int kt = 0; kt < 18; ++kt) {
        const int dydx = kt >> 1;
        const int dy = dydx / 3, dx = dydx - 3 * dy;
        const int coff = (kt & 1) * 32;
        // A fragment (16x32 f16): lane lo = pixel row M; runs of 8 c_in.
        const int abase = (dy * 66 + (wave * 16 + lo + dx)) * CC + coff + hi * 8;
        v8h r0 = *(const v8h*)&s_in[abase];
        v8h r1 = *(const v8h*)&s_in[abase + 16];
        v16h afrag;
        #pragma unroll
        for (int i = 0; i < 8; ++i) { afrag[i] = r0[i]; afrag[8 + i] = r1[i]; }
        // B fragments: pre-swizzled, lane-coalesced 32B loads from L2-resident buf.
        const _Float16* wb = wfrag + (((size_t)kt * 4) * 32 + lane) * 16;
        v16h b0 = *(const v16h*)(wb);
        v16h b1 = *(const v16h*)(wb + 32 * 16);
        v16h b2 = *(const v16h*)(wb + 2 * 32 * 16);
        v16h b3 = *(const v16h*)(wb + 3 * 32 * 16);
        acc0 = __builtin_amdgcn_wmma_f32_16x16x32_f16(false, afrag, false, b0, (short)0, acc0, false, false);
        acc1 = __builtin_amdgcn_wmma_f32_16x16x32_f16(false, afrag, false, b1, (short)0, acc1, false, false);
        acc2 = __builtin_amdgcn_wmma_f32_16x16x32_f16(false, afrag, false, b2, (short)0, acc2, false, false);
        acc3 = __builtin_amdgcn_wmma_f32_16x16x32_f16(false, afrag, false, b3, (short)0, acc3, false, false);
    }

    // Epilogue: bias + relu, optional store, mask-conv partial via xor-shuffle.
    const float bi0 = bias[lo],      bi1 = bias[16 + lo];
    const float bi2 = bias[32 + lo], bi3 = bias[48 + lo];
    const float wm0 = wmask[wm_off + lo],      wm1 = wmask[wm_off + 16 + lo];
    const float wm2 = wmask[wm_off + 32 + lo], wm3 = wmask[wm_off + 48 + lo];
    float psum[8];
    #pragma unroll
    for (int r = 0; r < 8; ++r) {
        float v0 = fmaxf(acc0[r] + bi0, 0.f);
        float v1 = fmaxf(acc1[r] + bi1, 0.f);
        float v2 = fmaxf(acc2[r] + bi2, 0.f);
        float v3 = fmaxf(acc3[r] + bi3, 0.f);
        psum[r] = wm0 * v0 + wm1 * v1 + wm2 * v2 + wm3 * v3;
        if (maux_out) {
            // C layout: element r -> pixel m = r + 8*hi, channel n = lo (+cbase)
            int x = xbase + wave * 16 + r + 8 * hi;
            size_t p = (((size_t)b * CC + lo) * HH + y) * WW + x;
            const size_t cs = (size_t)16 * HH * WW;
            maux_out[p]          = (_Float16)v0;
            maux_out[p + cs]     = (_Float16)v1;
            maux_out[p + 2 * cs] = (_Float16)v2;
            maux_out[p + 3 * cs] = (_Float16)v3;
        }
    }
    #pragma unroll
    for (int r = 0; r < 8; ++r) {
        float s = psum[r];
        for (int off = 1; off < 16; off <<= 1) s += __shfl_xor(s, off, 32);
        if (lo == 0) {
            int x = xbase + wave * 16 + r + 8 * hi;
            partial_out[((size_t)b * HH + y) * WW + x] = s;
        }
    }
}

// ---------------------------------------------------------------------------
// Kernel 3: mask = sigmoid(pm + pa + b_mask); out = m + mask * m_aux
// ---------------------------------------------------------------------------
__global__ __launch_bounds__(256)
void finalize_kernel(const float* __restrict__ m, const _Float16* __restrict__ maux,
                     const float* __restrict__ pm, const float* __restrict__ pa,
                     const float* __restrict__ bmask, float* __restrict__ out) {
    size_t idx = (size_t)blockIdx.x * 256 + threadIdx.x;
    const size_t HW = (size_t)HH * WW, CHW = (size_t)CC * HW;
    if (idx >= (size_t)BB * CHW) return;
    size_t b   = idx / CHW;
    size_t pix = idx % HW;
    float p = pm[b * HW + pix] + pa[b * HW + pix] + bmask[0];
    float mask = 1.0f / (1.0f + expf(-p));
    out[idx] = m[idx] + mask * (float)maux[idx];
}

// ---------------------------------------------------------------------------
extern "C" void kernel_launch(void* const* d_in, const int* in_sizes, int n_in,
                              void* d_out, int out_size, void* d_ws, size_t ws_size,
                              hipStream_t stream) {
    (void)in_sizes; (void)n_in; (void)out_size; (void)ws_size;
    // Input order: m, f, r, b, l, u, d, w_fusion, b_fusion, w_mask, b_mask
    const float* m   = (const float*)d_in[0];
    const float* ff  = (const float*)d_in[1];
    const float* fr  = (const float*)d_in[2];
    const float* fb  = (const float*)d_in[3];
    const float* fl  = (const float*)d_in[4];
    const float* fu  = (const float*)d_in[5];
    const float* fd  = (const float*)d_in[6];
    const float* wfu = (const float*)d_in[7];
    const float* bfu = (const float*)d_in[8];
    const float* wmk = (const float*)d_in[9];
    const float* bmk = (const float*)d_in[10];
    float* out = (float*)d_out;

    // Workspace carve-up (~33.3 MB total)
    char* ws = (char*)d_ws;
    _Float16* aux   = (_Float16*)(ws);                         // 16 MiB
    _Float16* maux  = (_Float16*)(ws + 16777216);              // 16 MiB
    float*    pm    = (float*)   (ws + 33554432);              // 512 KiB
    float*    pa    = (float*)   (ws + 34078720);              // 512 KiB
    _Float16* wfrag = (_Float16*)(ws + 34603008);              // 72 KiB

    // 0) weight swizzle to WMMA B-fragment order
    prep_weights<<<144, 256, 0, stream>>>(wfu, wfrag);
    // 1) cube -> equirect (f16 aux)
    c2e_kernel<<<dim3(HH, 16), 256, 0, stream>>>(fb, fd, ff, fl, fr, fu, aux);
    // 2a) main conv: only mask partial (m_main never materialized)
    conv_fusion_wmma<float><<<dim3(4, HH, BB), 128, 0, stream>>>(
        m, wfrag, bfu, wmk, 0, ((_Float16*)nullptr), pm);
    // 2b) aux conv: m_aux (f16) + mask partial
    conv_fusion_wmma<_Float16><<<dim3(4, HH, BB), 128, 0, stream>>>(
        aux, wfrag, bfu, wmk, 64, maux, pa);
    // 3) sigmoid mask + residual blend
    finalize_kernel<<<(BB * CC * HH * WW + 255) / 256, 256, 0, stream>>>(
        m, maux, pm, pa, bmk, out);
}
